// SparseMoE_44873818308994
// MI455X (gfx1250) — compile-verified, matrix operand order checked by reading, not run on hardware
//
#include <hip/hip_runtime.h>

typedef __attribute__((ext_vector_type(16))) __bf16          v16bf;
typedef __attribute__((ext_vector_type(8)))  float           v8f;
typedef __attribute__((ext_vector_type(4)))  unsigned int    u32x4;
typedef __attribute__((ext_vector_type(4)))  int             i32x4;
typedef __attribute__((ext_vector_type(4)))  float           f32x4;
typedef __attribute__((ext_vector_type(4)))  unsigned short  u16x4;

#define AS1 __attribute__((address_space(1)))
#define AS3 __attribute__((address_space(3)))

struct U32x8 { u32x4 lo, hi; };

constexpr int Bc = 8, Sc = 2048, Tt = Bc * Sc;   // 16384 tokens
constexpr int Dd = 512, Ee = 10, Hh = 2048;
constexpr int BM = 128;                          // tokens per block tile
constexpr int BH = 64;                           // hidden chunk
constexpr int TILES_PER_E = Tt / BM;             // 128

constexpr int SXS = 520;                         // sX / sW1 row stride (bf16), padded
constexpr int SWS = 72;                          // sW2 / sH row stride (bf16), padded

constexpr int OFF_X  = 0;                                // [128][520] bf16
constexpr int OFF_W1 = OFF_X  + BM * SXS * 2;            // 133120, [64][520] bf16
constexpr int OFF_W2 = OFF_W1 + 64 * SXS * 2;            // 199680, [512][72] bf16
constexpr int OFF_H  = OFF_W2 + Dd * SWS * 2;            // 273408, [128][72] bf16
constexpr int OFF_T  = OFF_H  + BM * SWS * 2;            // 291840
constexpr int OFF_G  = OFF_T  + BM * 4;                  // 292352
constexpr int SMEM_BYTES = OFF_G + BM * 4;               // 292864 (< 320K)

__device__ __forceinline__ unsigned short f2bf(float f) {
  unsigned int u = __builtin_bit_cast(unsigned int, f);
  u += 0x7fffu + ((u >> 16) & 1u);               // round-to-nearest-even
  return (unsigned short)(u >> 16);
}

__device__ __forceinline__ v16bf ld_frag(const unsigned short* p) {
  U32x8 r;
  r.lo = *(const u32x4*)(p);
  r.hi = *(const u32x4*)(p + 16);
  return __builtin_bit_cast(v16bf, r);
}

__device__ __forceinline__ v8f wmma_bf16(v16bf a, v16bf b, v8f c) {
  return __builtin_amdgcn_wmma_f32_16x16x32_bf16(false, a, false, b,
                                                 (short)0, c, false, false);
}

// ---- CDNA5 async global->LDS copy (ASYNCcnt) with portable fallback ----
#if __has_builtin(__builtin_amdgcn_global_load_async_to_lds_b128) && \
    __has_builtin(__builtin_amdgcn_s_wait_asynccnt)
#define HAVE_ASYNC_LDS 1
__device__ __forceinline__ void async_cp16(const unsigned short* g, unsigned short* l) {
  i32x4* gv = reinterpret_cast<i32x4*>(const_cast<unsigned short*>(g));
  i32x4* lv = reinterpret_cast<i32x4*>(l);
  __builtin_amdgcn_global_load_async_to_lds_b128(
      (AS1 i32x4*)gv, (AS3 i32x4*)lv, 0, 0);
}
#define ASYNC_WAIT(n) __builtin_amdgcn_s_wait_asynccnt(n)
#else
#define HAVE_ASYNC_LDS 0
__device__ __forceinline__ void async_cp16(const unsigned short* g, unsigned short* l) {
  *(u32x4*)l = *(const u32x4*)g;   // sync fallback; barriers provide ordering
}
#define ASYNC_WAIT(n) ((void)0)
#endif

// ---------------- weight transpose + fp32 -> bf16 ----------------
// src: [E][M][N] fp32 row-major, dst: [E][N][M] bf16
__global__ void transpose_cvt_kernel(const float* __restrict__ src,
                                     unsigned short* __restrict__ dst,
                                     int M, int N) {
  __shared__ float tile[32][33];
  int e = blockIdx.z;
  const float* s = src + (size_t)e * M * N;
  unsigned short* d = dst + (size_t)e * M * N;
  int r0 = blockIdx.x * 32;   // along M
  int c0 = blockIdx.y * 32;   // along N
  int tx = threadIdx.x & 31;
  int ty = threadIdx.x >> 5;  // 0..7
#pragma unroll
  for (int i = 0; i < 4; ++i) {
    int r = ty + i * 8;
    tile[r][tx] = s[(size_t)(r0 + r) * N + c0 + tx];
  }
  __syncthreads();
#pragma unroll
  for (int i = 0; i < 4; ++i) {
    int r = ty + i * 8;
    d[(size_t)(c0 + r) * M + r0 + tx] = f2bf(tile[tx][r]);
  }
}

// ---------------- fp32 noisy top-2 router ----------------
__global__ void router_kernel(const float* __restrict__ x,
                              const float* __restrict__ noise,
                              const float* __restrict__ Wr,
                              const float* __restrict__ br,
                              const float* __restrict__ Wn,
                              const float* __restrict__ bn,
                              int* __restrict__ counts,
                              int* __restrict__ btok,
                              float* __restrict__ bgate) {
  int t = blockIdx.x * blockDim.x + threadIdx.x;
  if (t >= Tt) return;
  float accr[Ee], accn[Ee];
#pragma unroll
  for (int e = 0; e < Ee; ++e) { accr[e] = br[e]; accn[e] = bn[e]; }
  const float* xr = x + (size_t)t * Dd;
  for (int k = 0; k < Dd; ++k) {
    float xv = xr[k];
    const float* wr = Wr + k * Ee;
    const float* wn = Wn + k * Ee;
#pragma unroll
    for (int e = 0; e < Ee; ++e) {
      accr[e] = fmaf(xv, wr[e], accr[e]);
      accn[e] = fmaf(xv, wn[e], accn[e]);
    }
  }
  float nv[Ee];
#pragma unroll
  for (int e = 0; e < Ee; ++e) {
    float v = accn[e];
    float sp = fmaxf(v, 0.f) + log1pf(expf(-fabsf(v)));  // softplus
    nv[e] = accr[e] + noise[(size_t)t * Ee + e] * sp;
  }
  int i0 = 0; float v0 = nv[0];
#pragma unroll
  for (int e = 1; e < Ee; ++e) if (nv[e] > v0) { v0 = nv[e]; i0 = e; }
  int i1 = (i0 == 0) ? 1 : 0; float v1 = nv[i1];
#pragma unroll
  for (int e = 0; e < Ee; ++e) if (e != i0 && nv[e] > v1) { v1 = nv[e]; i1 = e; }
  float g0 = 1.f / (1.f + expf(v1 - v0));
  float g1 = 1.f - g0;
  int s0 = atomicAdd(&counts[i0], 1);
  btok[i0 * Tt + s0] = t; bgate[i0 * Tt + s0] = g0;
  int s1 = atomicAdd(&counts[i1], 1);
  btok[i1 * Tt + s1] = t; bgate[i1 * Tt + s1] = g1;
}

// ---------------- fused expert FFN: relu(x@W1+b1)@W2+b2, gated scatter-add ----------------
__global__ __launch_bounds__(512) void moe_ffn_kernel(
    const float* __restrict__ x,
    const unsigned short* __restrict__ W1T,   // [E][H][D] bf16
    const float* __restrict__ pb1,            // [E][H]
    const unsigned short* __restrict__ W2T,   // [E][D][H] bf16
    const float* __restrict__ pb2,            // [E][D]
    const int* __restrict__ counts,
    const int* __restrict__ btok,
    const float* __restrict__ bgate,
    float* __restrict__ out) {
  int e  = blockIdx.x >> 7;        // TILES_PER_E == 128
  int mt = blockIdx.x & 127;
  int cnt = counts[e];
  int m0 = mt * BM;
  if (m0 >= cnt) return;           // uniform early exit

  extern __shared__ char smem[];
  unsigned short* sX  = (unsigned short*)(smem + OFF_X);   // [128][520]
  unsigned short* sW1 = (unsigned short*)(smem + OFF_W1);  // [64][520]
  unsigned short* sW2 = (unsigned short*)(smem + OFF_W2);  // [512][72]
  unsigned short* sH  = (unsigned short*)(smem + OFF_H);   // [128][72]
  int*   sT = (int*)(smem + OFF_T);
  float* sG = (float*)(smem + OFF_G);

  int tid = threadIdx.x;
  const unsigned short* W1e = W1T + (size_t)e * Hh * Dd;
  const unsigned short* W2e = W2T + (size_t)e * Dd * Hh;

  // --- prefetch W1 chunk 0 (async, overlaps the X gather) ---
#pragma unroll
  for (int it = 0; it < 8; ++it) {
    int idx  = it * 512 + tid;
    int hrow = idx >> 6;             // 0..63
    int sub  = idx & 63;             // 64 x 8 elems per row
    async_cp16(W1e + (size_t)hrow * Dd + sub * 8, sW1 + hrow * SXS + sub * 8);
  }

  if (tid < BM) {
    int slot = m0 + tid;
    int tk = 0; float g = 0.f;
    if (slot < cnt) { tk = btok[e * Tt + slot]; g = bgate[e * Tt + slot]; }
    sT[tid] = tk; sG[tid] = g;
  }
  __syncthreads();

  {  // gather 128 token rows of x, convert to bf16 into LDS
    int row = tid >> 2;            // 4 threads per row
    int kc  = (tid & 3) * 128;
    const float* xr = x + (size_t)sT[row] * Dd + kc;
    unsigned short* dr = sX + row * SXS + kc;
#pragma unroll
    for (int i = 0; i < 128; i += 4) {
      f32x4 v = *(const f32x4*)(xr + i);
      u16x4 b;
      b.x = f2bf(v.x); b.y = f2bf(v.y); b.z = f2bf(v.z); b.w = f2bf(v.w);
      *(u16x4*)(dr + i) = b;
    }
  }

  int lane = tid & 31;
  int w    = tid >> 5;   // 16 waves
  int hf   = lane >> 4;  // lane half (K-offset group)
  int l16  = lane & 15;
  int rp   = w & 3;      // row pair: rows rp*32 + {0..15, 16..31}
  int cq   = w >> 2;     // 0..3: stage-1 col tile cq*16; stage-2 cols cq*128..+127

  v8f o0[8], o1[8];
#pragma unroll
  for (int j = 0; j < 8; ++j) { o0[j] = (v8f){}; o1[j] = (v8f){}; }

  const unsigned short* aX0 = sX + (rp * 32 + l16) * SXS + hf * 8;
  const unsigned short* aX1 = aX0 + 16 * SXS;

#pragma unroll 1
  for (int h0 = 0; h0 < Hh; h0 += BH) {
    ASYNC_WAIT(0);                 // W1 panel for this chunk resident; sW2 free
    __syncthreads();

    // issue async W2 panel for this chunk (overlaps stage 1)
#pragma unroll
    for (int it = 0; it < 8; ++it) {
      int idx = it * 512 + tid;
      int d   = idx >> 3;            // 0..511
      int sub = idx & 7;             // 8 x 8 elems per row
      async_cp16(W2e + (size_t)d * Hh + h0 + sub * 8, sW2 + d * SWS + sub * 8);
    }

    // ---- stage 1: hidden[128][64] = sX @ W1 panel, 2 row tiles per wave ----
    const unsigned short* bP = sW1 + (cq * 16 + l16) * SXS + hf * 8;
    v8f h0a = (v8f){}, h1a = (v8f){};
    v16bf bC = ld_frag(bP);
#pragma unroll
    for (int kk = 0; kk < Dd; kk += 32) {
      v16bf a0 = ld_frag(aX0 + kk);
      v16bf a1 = ld_frag(aX1 + kk);
      v16bf bN = bC;
      if (kk + 32 < Dd) bN = ld_frag(bP + kk + 32);
      h0a = wmma_bf16(a0, bC, h0a);
      h1a = wmma_bf16(a1, bC, h1a);
      bC = bN;
    }
    int hc = h0 + cq * 16 + l16;
    float bias1 = pb1[e * Hh + hc];
#pragma unroll
    for (int v = 0; v < 8; ++v) {
      int m = rp * 32 + hf * 8 + v;
      sH[m * SWS + cq * 16 + l16]        = f2bf(fmaxf(h0a[v] + bias1, 0.f));
      sH[(m + 16) * SWS + cq * 16 + l16] = f2bf(fmaxf(h1a[v] + bias1, 0.f));
    }
    ASYNC_WAIT(0);                 // W2 panel resident
    __syncthreads();               // sH + sW2 visible; sW1 now idle

    // prefetch next W1 panel (overlaps stage 2)
    bool more = (h0 + BH) < Hh;
    if (more) {
#pragma unroll
      for (int it = 0; it < 8; ++it) {
        int idx  = it * 512 + tid;
        int hrow = idx >> 6;
        int sub  = idx & 63;
        async_cp16(W1e + (size_t)(h0 + BH + hrow) * Dd + sub * 8,
                   sW1 + hrow * SXS + sub * 8);
      }
    }

    // ---- stage 2: out[128][512] += sH @ sW2 panel, 2x8 tiles per wave ----
    const unsigned short* b2B = sW2 + (cq * 128 + l16) * SWS + hf * 8;
    const unsigned short* aH0 = sH + (rp * 32 + l16) * SWS + hf * 8;
    const unsigned short* aH1 = aH0 + 16 * SWS;
#pragma unroll
    for (int kk2 = 0; kk2 < BH; kk2 += 32) {
      v16bf a20 = ld_frag(aH0 + kk2);
      v16bf a21 = ld_frag(aH1 + kk2);
      v16bf b2C = ld_frag(b2B + kk2);
#pragma unroll
      for (int j = 0; j < 8; ++j) {
        v16bf b2N = b2C;
        if (j < 7) b2N = ld_frag(b2B + (j + 1) * 16 * SWS + kk2);
        o0[j] = wmma_bf16(a20, b2C, o0[j]);
        o1[j] = wmma_bf16(a21, b2C, o1[j]);
        b2C = b2N;
      }
    }
  }

  // ---- epilogue: (acc + b2) * gate, scatter-add per token (tok/gate from LDS) ----
#pragma unroll
  for (int j = 0; j < 8; ++j) {
    int dcol = cq * 128 + j * 16 + l16;
    float bias2 = pb2[e * Dd + dcol];
#pragma unroll
    for (int v = 0; v < 8; ++v) {
      int m = rp * 32 + hf * 8 + v;
      atomicAdd(out + (size_t)sT[m] * Dd + dcol,        (o0[j][v] + bias2) * sG[m]);
      atomicAdd(out + (size_t)sT[m + 16] * Dd + dcol,   (o1[j][v] + bias2) * sG[m + 16]);
    }
  }
}

// ---------------- host launcher ----------------
extern "C" void kernel_launch(void* const* d_in, const int* in_sizes, int n_in,
                              void* d_out, int out_size, void* d_ws, size_t ws_size,
                              hipStream_t stream) {
  const float* x     = (const float*)d_in[0];
  const float* noise = (const float*)d_in[1];
  const float* Wr    = (const float*)d_in[2];
  const float* br    = (const float*)d_in[3];
  const float* Wn    = (const float*)d_in[4];
  const float* bn    = (const float*)d_in[5];
  const float* W1    = (const float*)d_in[6];
  const float* b1    = (const float*)d_in[7];
  const float* W2    = (const float*)d_in[8];
  const float* b2    = (const float*)d_in[9];
  float* out = (float*)d_out;
  (void)in_sizes; (void)n_in; (void)out_size; (void)ws_size;

  char* ws = (char*)d_ws;
  size_t o = 0;
  auto alloc = [&](size_t bytes) {
    o = (o + 255) & ~(size_t)255;
    char* p = ws + o;
    o += bytes;
    return p;
  };
  int*   counts = (int*)  alloc(Ee * sizeof(int));
  int*   btok   = (int*)  alloc((size_t)Ee * Tt * sizeof(int));
  float* bgate  = (float*)alloc((size_t)Ee * Tt * sizeof(float));
  unsigned short* W1T = (unsigned short*)alloc((size_t)Ee * Hh * Dd * 2);
  unsigned short* W2T = (unsigned short*)alloc((size_t)Ee * Hh * Dd * 2);

  (void)hipFuncSetAttribute((const void*)moe_ffn_kernel,
                            hipFuncAttributeMaxDynamicSharedMemorySize, SMEM_BYTES);

  (void)hipMemsetAsync(counts, 0, Ee * sizeof(int), stream);
  (void)hipMemsetAsync(out, 0, (size_t)Tt * Dd * sizeof(float), stream);

  transpose_cvt_kernel<<<dim3(Dd / 32, Hh / 32, Ee), 256, 0, stream>>>(W1, W1T, Dd, Hh);
  transpose_cvt_kernel<<<dim3(Hh / 32, Dd / 32, Ee), 256, 0, stream>>>(W2, W2T, Hh, Dd);
  router_kernel<<<Tt / 256, 256, 0, stream>>>(x, noise, Wr, br, Wn, bn,
                                              counts, btok, bgate);
  moe_ffn_kernel<<<dim3(Ee * TILES_PER_E), dim3(512), SMEM_BYTES, stream>>>(
      x, W1T, b1, W2T, b2, counts, btok, bgate, out);
}
